// ElementalModesMessagePassingNeuralNetwork_3573412790695
// MI455X (gfx1250) — matrix operand order repaired and checked by reading, not compiled, not run on hardware
//
#include <hip/hip_runtime.h>

// ============================================================================
// PhysNet-style MPNN forward for MI455X (gfx1250, wave32, WMMA bf16->f32).
// v2: all WMMA inputs staged in LDS as bf16 in A-fragment order, so each
// element is converted f32->bf16 exactly once and A-frags are single
// contiguous 32B ds loads (kills the VALU-heavy per-use conversion of v1).
//
// d_in pytree-flat order (jax sorted-dict flatten):
//   0 Z, 1 R, 2 idx_i, 3 idx_j, 4 M, 5 QaAlpha, 6 QaBeta, 7 batch_seg,
//   8 params.W (5,2), 9 params.b (2),
//   10..179 blocks[5] x 34 leaves:
//     +0 G, +1 di.W +2 di.b, +3 dj.W +4 dj.b, +5 do.W +6 do.b, +7 ds.W +8 ds.b,
//     +9..12 out_res[0]{d1.W,d1.b,d2.W,d2.b}, +13..20 res_at[0..1]{...},
//     +21..32 res_int[0..2]{...}, +33 u
//   180 centers, 181..186 em_hidden[0].{W,b}, em_hidden[1].{W,b}, em_out.{W,b},
//   187 scales (2,95), 188 shifts (2,95), 189 widths (scalar)
//
// d_out: [N*2 outputs | E Dij | 1 nhloss] floats.
// ============================================================================

#define NATOMS 20000
#define NEDGES 640000
#define FDIM   128
#define KRBF   64
#define NBLOCKS 5
#define SRCUT  8.0f

typedef __attribute__((ext_vector_type(16))) __bf16 v16bf;
typedef __attribute__((ext_vector_type(8)))  float  v8f;

// ---------------------------------------------------------------- helpers ---
__device__ __forceinline__ __bf16 f2bf(float f) {
  unsigned u = __builtin_bit_cast(unsigned, f);
  unsigned r = (u + 0x7FFFu + ((u >> 16) & 1u)) >> 16;   // RNE
  unsigned short h = (unsigned short)r;
  return __builtin_bit_cast(__bf16, h);
}

__device__ __forceinline__ float sspf(float x) {         // softplus(x) - ln2
  float ax = fabsf(x);
  return fmaxf(x, 0.0f) + log1pf(__expf(-ax)) - 0.69314718055994530942f;
}

// Pre-packed B fragment: 32 lanes x 16 contiguous bf16 per (K-tile, N-tile).
__device__ __forceinline__ v16bf b_frag(const __bf16* wf, int bi) {
  int lane = threadIdx.x & 31;
  return *(const v16bf*)(wf + ((size_t)bi * 32 + lane) * 16);
}

// Dense over K=128 from an LDS bf16 buffer already in A-fragment order
// ([4 ktiles][32 lanes][16 elems]). One 16x16 f32 tile per wave, 4 WMMAs.
__device__ __forceinline__ v8f dense_tile_bf(const __bf16* Tb, const __bf16* wf, int nt) {
  int lane = threadIdx.x & 31;
  v8f acc = {0.f, 0.f, 0.f, 0.f, 0.f, 0.f, 0.f, 0.f};
#pragma unroll
  for (int kt = 0; kt < 4; ++kt) {
    v16bf a = *(const v16bf*)(Tb + kt * 512 + lane * 16);
    v16bf b = b_frag(wf, kt * 8 + nt);
    acc = __builtin_amdgcn_wmma_f32_16x16x32_bf16(false, a, false, b, (short)0, acc,
                                                  false, false);
  }
  return acc;
}

// Elementwise pass: row-major f32 LDS tile [16][128] -> bf16 A-frag order,
// with ssp applied. Forward K map (ISA 7.12.2): k = (e<8?e:e+8) + 8*half.
__device__ __forceinline__ void make_frag_ssp(__bf16* dst, const float* src) {
  for (int i = threadIdx.x; i < 2048; i += 256) {
    int kt = i >> 9, rem = i & 511, lane_r = rem >> 4, e = rem & 15;
    int m = lane_r & 15, half = lane_r >> 4;
    int k = kt * 32 + ((e < 8) ? e : e + 8) + 8 * half;
    dst[i] = f2bf(sspf(src[m * FDIM + k]));
  }
}

// Wave writes its 16x16 C-tile column (col n = K index of the NEXT dense)
// directly into a bf16 A-frag-order buffer, applying bias + ssp.
// Inverse K map: half_r=(kq>>3)&1, e_r=(kq&7)|(kq>=16?8:0).
__device__ __forceinline__ void write_frag_col(__bf16* dst, v8f t, float b,
                                               int n, int half_w) {
  int kt = n >> 5, kq = n & 31;
  int half_r = (kq >> 3) & 1;
  int e_r = (kq & 7) | (kq >= 16 ? 8 : 0);
  __bf16* base = dst + kt * 512 + half_r * 256 + e_r;
#pragma unroll
  for (int rr = 0; rr < 8; ++rr) {
    int m = rr + 8 * half_w;
    base[m * 16] = f2bf(sspf(t[rr] + b));
  }
}

// -------------------------------------------------------- weight repacking ---
// W is f32 [Kdim,Ndim] row-major. Dense-B striping: elem e -> K = (e<8?e:e+8) + 8*half.
__global__ void k_prep_bfrag(const float* __restrict__ W, __bf16* __restrict__ frag,
                             int Kdim, int Ndim) {
  int bi = blockIdx.x, lane = threadIdx.x;
  int ntiles = Ndim >> 4;
  int kt = bi / ntiles, nt = bi - kt * ntiles;
  int col = nt * 16 + (lane & 15), half = lane >> 4;
  __bf16* dst = frag + ((size_t)bi * 32 + lane) * 16;
#pragma unroll
  for (int e = 0; e < 16; ++e) {
    int k = kt * 32 + ((e < 8) ? e : e + 8) + 8 * half;
    dst[e] = f2bf(W[(size_t)k * Ndim + col]);
  }
}

// -------------------------------------------------------------------- init ---
__global__ void k_init(float* out_acc, float* last2, float* nh) {
  int i = blockIdx.x * 256 + threadIdx.x;
  if (i < NATOMS * 2) { out_acc[i] = 0.0f; last2[i] = 0.0f; }
  if (i == 0) nh[0] = 0.0f;
}

// --------------------------------------------------------------- distances ---
__global__ void k_dij(const float* __restrict__ R, const int* __restrict__ ii,
                      const int* __restrict__ jj, float* __restrict__ D) {
  int e = blockIdx.x * 256 + threadIdx.x;
  if (e >= NEDGES) return;
  int a = ii[e], b = jj[e];
  float dx = R[a * 3 + 0] - R[b * 3 + 0];
  float dy = R[a * 3 + 1] - R[b * 3 + 1];
  float dz = R[a * 3 + 2] - R[b * 3 + 2];
  float s = dx * dx + dy * dy + dz * dz;
  D[e] = sqrtf(fmaxf(s, 0.0f));
}

// ---------------------------------------------------------- embedding MLP ---
__global__ void k_embed(const int* __restrict__ Z, const float* __restrict__ M,
                        const float* __restrict__ Qa, const float* __restrict__ Qb,
                        const float* __restrict__ W0, const float* __restrict__ b0,
                        const float* __restrict__ W1, const float* __restrict__ b1,
                        const float* __restrict__ W2, const float* __restrict__ b2,
                        float* __restrict__ x) {
  __shared__ float h1[256], h2[256];
  int atom = blockIdx.x, t = threadIdx.x;
  float f0 = (float)Z[atom], f1 = M[atom], f2 = Qa[atom], f3 = Qb[atom];
  float s = b0[t] + f0 * W0[0 * 256 + t] + f1 * W0[1 * 256 + t] +
            f2 * W0[2 * 256 + t] + f3 * W0[3 * 256 + t];
  h1[t] = sspf(s);
  __syncthreads();
  float s2 = b1[t];
  for (int k = 0; k < 256; ++k) s2 += h1[k] * W1[k * 256 + t];
  h2[t] = sspf(s2);
  __syncthreads();
  if (t < FDIM) {
    float s3 = b2[t];
    for (int k = 0; k < 256; ++k) s3 += h2[k] * W2[k * FDIM + t];
    x[(size_t)atom * FDIM + t] = s3;
  }
}

// --------------------------------------------- per-block atom-side prelude ---
// m = ssp(di(ssp(x))), hj = ssp(dj(ssp(x))). 16 atoms per WG, 8 waves.
__global__ void k_atom_pre(const float* __restrict__ x,
                           const __bf16* __restrict__ wdi, const float* __restrict__ bdi,
                           const __bf16* __restrict__ wdj, const float* __restrict__ bdj,
                           float* __restrict__ mbuf, float* __restrict__ hj) {
  __shared__ alignas(32) __bf16 Tb[2048];
  int tid = threadIdx.x;
  size_t gbase = (size_t)blockIdx.x * 16 * FDIM;
  for (int i = tid; i < 2048; i += 256) {    // ssp(x) -> frag order, one cvt/elem
    int kt = i >> 9, rem = i & 511, lane_r = rem >> 4, e = rem & 15;
    int m = lane_r & 15, half = lane_r >> 4;
    int k = kt * 32 + ((e < 8) ? e : e + 8) + 8 * half;
    Tb[i] = f2bf(sspf(x[gbase + (size_t)m * FDIM + k]));
  }
  __syncthreads();
  int nt = tid >> 5, lane = tid & 31, half = lane >> 4, n = nt * 16 + (lane & 15);
  v8f hi = dense_tile_bf(Tb, wdi, nt);
  v8f hv = dense_tile_bf(Tb, wdj, nt);
  float bi = bdi[n], bj = bdj[n];
#pragma unroll
  for (int r = 0; r < 8; ++r) {
    size_t row = gbase + (size_t)(r + 8 * half) * FDIM + n;
    mbuf[row] = sspf(hi[r] + bi);
    hj[row]   = sspf(hv[r] + bj);
  }
}

// ------------------------------------------------------------- edge kernel ---
// 1 wave = 16 edges. rbf[16][64] computed straight into bf16 A-frag order,
// g = rbf @ G via 2 WMMAs per n-tile, gather hj[idx_j]*g, then f32 atomics
// segment-sum into mbuf[idx_i] (L2-resident).
__global__ void k_edge(const float* __restrict__ Dij,
                       const int* __restrict__ idx_i, const int* __restrict__ idx_j,
                       const float* __restrict__ hj, float* __restrict__ mbuf,
                       const __bf16* __restrict__ gfrag,
                       const float* __restrict__ centers,
                       const float* __restrict__ widthsp) {
  __shared__ alignas(32) __bf16 rbb[8][1024];   // 2 ktiles x 32 lanes x 16
  int wave = threadIdx.x >> 5, lane = threadIdx.x & 31;
  int e0 = (blockIdx.x * 8 + wave) * 16;
  float widths = widthsp[0];

  // lane covers edge er=lane%16, k-range kh..kh+31 (kh selects the ktile)
  int er = lane & 15;
  int kt = lane >> 4;
  int kh = kt * 32;
  float D = Dij[e0 + er];
  float xq = D * (1.0f / SRCUT);
  float fc = 0.0f;
  if (xq < 1.0f) {
    float x2 = xq * xq, x3 = x2 * xq;
    fc = 1.0f + x3 * (-10.0f + 15.0f * xq - 6.0f * x2);
  }
  float eD = __expf(-D);
  __bf16* rb = rbb[wave] + kt * 512;
#pragma unroll 8
  for (int j = 0; j < 32; ++j) {              // write in frag order
    int half_r = (j >> 3) & 1;
    int e_r = (j & 7) | (j >= 16 ? 8 : 0);
    float d = eD - centers[kh + j];
    rb[half_r * 256 + er * 16 + e_r] = f2bf(fc * __expf(-widths * d * d));
  }
  __syncthreads();

  v16bf a0 = *(const v16bf*)(rbb[wave] + lane * 16);
  v16bf a1 = *(const v16bf*)(rbb[wave] + 512 + lane * 16);
  int half = lane >> 4, nl = lane & 15;
  int ji[8], si[8];
#pragma unroll
  for (int r = 0; r < 8; ++r) {
    int e = e0 + r + 8 * half;
    ji[r] = idx_j[e];
    si[r] = idx_i[e];
  }
#pragma unroll
  for (int nt = 0; nt < 8; ++nt) {
    v8f acc = {0.f, 0.f, 0.f, 0.f, 0.f, 0.f, 0.f, 0.f};
    acc = __builtin_amdgcn_wmma_f32_16x16x32_bf16(false, a0, false, b_frag(gfrag, nt),
                                                  (short)0, acc, false, false);
    acc = __builtin_amdgcn_wmma_f32_16x16x32_bf16(false, a1, false, b_frag(gfrag, 8 + nt),
                                                  (short)0, acc, false, false);
    int n = nt * 16 + nl;
#pragma unroll
    for (int r = 0; r < 8; ++r) {
      float mij = hj[(size_t)ji[r] * FDIM + n] * acc[r];
      atomicAdd(&mbuf[(size_t)si[r] * FDIM + n], mij);
    }
  }
}

// -------------------------------------------- per-block atom-side epilogue ---
struct PostArgs {
  float* x; float* mbuf; float* out_acc; float* last2; float* nh;
  const __bf16* wblk;                 // frag slots: 0 di,1 dj,2 ds,3..8 res_int,
  const float* b_ri[3][2];            //             9..12 res_at, 13..14 out_res
  const float* b_ra[2][2];
  const float* b_or0; const float* b_or1;
  const float* bds; const float* u;
  const float* wdo; const float* bdo; // f32 head weights [128,2]
  const float* Wglob; const float* bglob;
  int blk;
};
#define SLOT(s) (p.wblk + (size_t)(s) * 16384)

__global__ void k_atom_post(PostArgs p) {
  __shared__ float X[16 * FDIM], A[16 * FDIM];
  __shared__ alignas(32) __bf16 Tb[2048], Yb[2048];
  int tid = threadIdx.x;
  int a0 = blockIdx.x * 16;
  size_t gbase = (size_t)a0 * FDIM;
  for (int i = tid; i < 16 * FDIM; i += 256) {
    X[i] = p.x[gbase + i];
    A[i] = p.mbuf[gbase + i];
  }
  __syncthreads();

  int wave = tid >> 5, lane = tid & 31;
  int nt = wave, half = lane >> 4, nl = lane & 15, n = nt * 16 + nl;

  // ---- res_int x3 on A (message) ----
#pragma unroll 1
  for (int r = 0; r < 3; ++r) {
    make_frag_ssp(Tb, A);
    __syncthreads();
    v8f t1 = dense_tile_bf(Tb, SLOT(3 + 2 * r), nt);
    write_frag_col(Yb, t1, p.b_ri[r][0][n], n, half);
    __syncthreads();
    v8f t2 = dense_tile_bf(Yb, SLOT(4 + 2 * r), nt);
    float b2 = p.b_ri[r][1][n];
#pragma unroll
    for (int rr = 0; rr < 8; ++rr) A[(rr + 8 * half) * FDIM + n] += t2[rr] + b2;
    __syncthreads();
  }

  // ---- x = u*x + ds(ssp(m)) ----
  make_frag_ssp(Tb, A);
  __syncthreads();
  {
    v8f td = dense_tile_bf(Tb, SLOT(2), nt);
    float bd = p.bds[n], uu = p.u[n];
#pragma unroll
    for (int rr = 0; rr < 8; ++rr) {
      int ii = (rr + 8 * half) * FDIM + n;
      X[ii] = uu * X[ii] + td[rr] + bd;
    }
    __syncthreads();
  }

  // ---- res_at x2 on X ----
#pragma unroll 1
  for (int r = 0; r < 2; ++r) {
    make_frag_ssp(Tb, X);
    __syncthreads();
    v8f t1 = dense_tile_bf(Tb, SLOT(9 + 2 * r), nt);
    write_frag_col(Yb, t1, p.b_ra[r][0][n], n, half);
    __syncthreads();
    v8f t2 = dense_tile_bf(Yb, SLOT(10 + 2 * r), nt);
    float b2 = p.b_ra[r][1][n];
#pragma unroll
    for (int rr = 0; rr < 8; ++rr) X[(rr + 8 * half) * FDIM + n] += t2[rr] + b2;
    __syncthreads();
  }

  // write next-block state, keep a copy for the output head
  for (int i = tid; i < 16 * FDIM; i += 256) {
    float v = X[i];
    p.x[gbase + i] = v;
    A[i] = v;
  }
  __syncthreads();

  // ---- out_res x1 on A ----
  make_frag_ssp(Tb, A);
  __syncthreads();
  {
    v8f t1 = dense_tile_bf(Tb, SLOT(13), nt);
    write_frag_col(Yb, t1, p.b_or0[n], n, half);
    __syncthreads();
    v8f t2 = dense_tile_bf(Yb, SLOT(14), nt);
    float b2 = p.b_or1[n];
#pragma unroll
    for (int rr = 0; rr < 8; ++rr) A[(rr + 8 * half) * FDIM + n] += t2[rr] + b2;
    __syncthreads();
  }

  // ---- output head [16,2] + nhloss (ssp applied inline) ----
  if (tid < 32) {
    int at = tid >> 1, o = tid & 1;
    float s = p.bdo[o];
    for (int k = 0; k < FDIM; ++k) s += sspf(A[at * FDIM + k]) * p.wdo[k * 2 + o];
    size_t gi = (size_t)(a0 + at) * 2 + o;
    p.out_acc[gi] += s * p.Wglob[o] + p.bglob[o];
    float o2 = s * s;
    if (p.blk > 0) {
      float c = o2 / (o2 + p.last2[gi] + 1e-7f);
      atomicAdd(p.nh, c * (1.0f / (float)(NATOMS * 2)));
    }
    p.last2[gi] = o2;
  }
}

// ---------------------------------------------------------------- finalize ---
__global__ void k_finalize(const float* __restrict__ out_acc, const int* __restrict__ Z,
                           const float* __restrict__ scales, const float* __restrict__ shifts,
                           const float* __restrict__ nh, float* __restrict__ out) {
  int i = blockIdx.x * 256 + threadIdx.x;
  if (i < NATOMS * 2) {
    int atom = i >> 1, o = i & 1;
    int z = Z[atom];
    out[i] = out_acc[i] * scales[o * 95 + z] + shifts[o * 95 + z];
  }
  if (i == 0) out[NATOMS * 2 + NEDGES] = nh[0];
}

// ======================================================================= host
extern "C" void kernel_launch(void* const* d_in, const int* in_sizes, int n_in,
                              void* d_out_v, int out_size, void* d_ws, size_t ws_size,
                              hipStream_t stream) {
  (void)in_sizes; (void)n_in; (void)out_size; (void)ws_size;
  float* d_out = (float*)d_out_v;

  const int*   Z  = (const int*)d_in[0];
  const float* R  = (const float*)d_in[1];
  const int*   II = (const int*)d_in[2];
  const int*   JJ = (const int*)d_in[3];
  const float* Mw = (const float*)d_in[4];
  const float* Qa = (const float*)d_in[5];
  const float* Qb = (const float*)d_in[6];
  const float* PW = (const float*)d_in[8];
  const float* Pb = (const float*)d_in[9];
  const float* centers = (const float*)d_in[180];
  const float* scales  = (const float*)d_in[187];
  const float* shifts  = (const float*)d_in[188];
  const float* widths  = (const float*)d_in[189];

  // ---- workspace layout (bytes) ----
  char* ws = (char*)d_ws;
  const size_t WS_X    = 0;
  const size_t WS_HJ   = WS_X    + (size_t)NATOMS * FDIM * 4;
  const size_t WS_M    = WS_HJ   + (size_t)NATOMS * FDIM * 4;
  const size_t WS_OUT  = WS_M    + (size_t)NATOMS * FDIM * 4;
  const size_t WS_LAST = WS_OUT  + (size_t)NATOMS * 2 * 4;
  const size_t WS_NH   = WS_LAST + (size_t)NATOMS * 2 * 4;
  const size_t WS_FRAG = WS_NH   + 256;
  const size_t BLK_FRAG = (size_t)15 * 16384 + 8192;   // bf16 elems per block

  float* xbuf    = (float*)(ws + WS_X);
  float* hjbuf   = (float*)(ws + WS_HJ);
  float* mbuf    = (float*)(ws + WS_M);
  float* out_acc = (float*)(ws + WS_OUT);
  float* last2   = (float*)(ws + WS_LAST);
  float* nh      = (float*)(ws + WS_NH);

  // ---- repack weights to bf16 WMMA B-fragments (every call; deterministic) ----
  static const int slot_leaf[15] = {1, 3, 7,                // di.W, dj.W, ds.W
                                    21, 23, 25, 27, 29, 31, // res_int d1/d2 x3
                                    13, 15, 17, 19,         // res_at d1/d2 x2
                                    9, 11};                 // out_res d1/d2
  for (int b = 0; b < NBLOCKS; ++b) {
    int base = 10 + b * 34;
    __bf16* fb = (__bf16*)(ws + WS_FRAG) + (size_t)b * BLK_FRAG;
    for (int s = 0; s < 15; ++s)
      k_prep_bfrag<<<32, 32, 0, stream>>>((const float*)d_in[base + slot_leaf[s]],
                                          fb + (size_t)s * 16384, FDIM, FDIM);
    k_prep_bfrag<<<16, 32, 0, stream>>>((const float*)d_in[base + 0],
                                        fb + (size_t)15 * 16384, KRBF, FDIM);
  }

  k_init<<<(NATOMS * 2 + 255) / 256, 256, 0, stream>>>(out_acc, last2, nh);
  k_dij<<<(NEDGES + 255) / 256, 256, 0, stream>>>(R, II, JJ, d_out + NATOMS * 2);
  k_embed<<<NATOMS, 256, 0, stream>>>(Z, Mw, Qa, Qb,
      (const float*)d_in[181], (const float*)d_in[182],
      (const float*)d_in[183], (const float*)d_in[184],
      (const float*)d_in[185], (const float*)d_in[186], xbuf);

  for (int b = 0; b < NBLOCKS; ++b) {
    int base = 10 + b * 34;
    __bf16* fb = (__bf16*)(ws + WS_FRAG) + (size_t)b * BLK_FRAG;

    k_atom_pre<<<NATOMS / 16, 256, 0, stream>>>(
        xbuf, fb, (const float*)d_in[base + 2],
        fb + 16384, (const float*)d_in[base + 4], mbuf, hjbuf);

    k_edge<<<NEDGES / 128, 256, 0, stream>>>(
        d_out + NATOMS * 2, II, JJ, hjbuf, mbuf,
        fb + (size_t)15 * 16384, centers, widths);

    PostArgs pa;
    pa.x = xbuf; pa.mbuf = mbuf; pa.out_acc = out_acc; pa.last2 = last2; pa.nh = nh;
    pa.wblk = fb;
    for (int r = 0; r < 3; ++r) {
      pa.b_ri[r][0] = (const float*)d_in[base + 22 + 4 * r];
      pa.b_ri[r][1] = (const float*)d_in[base + 24 + 4 * r];
    }
    for (int r = 0; r < 2; ++r) {
      pa.b_ra[r][0] = (const float*)d_in[base + 14 + 4 * r];
      pa.b_ra[r][1] = (const float*)d_in[base + 16 + 4 * r];
    }
    pa.b_or0 = (const float*)d_in[base + 10];
    pa.b_or1 = (const float*)d_in[base + 12];
    pa.bds = (const float*)d_in[base + 8];
    pa.u   = (const float*)d_in[base + 33];
    pa.wdo = (const float*)d_in[base + 5];
    pa.bdo = (const float*)d_in[base + 6];
    pa.Wglob = PW + b * 2;
    pa.bglob = Pb;
    pa.blk = b;
    k_atom_post<<<NATOMS / 16, 256, 0, stream>>>(pa);
  }

  k_finalize<<<(NATOMS * 2 + 255) / 256, 256, 0, stream>>>(out_acc, Z, scales, shifts,
                                                           nh, d_out);
}
